// TransformerDecoder_43190191128976
// MI455X (gfx1250) — matrix operand order repaired and checked by reading, not compiled, not run on hardware
//
#include <hip/hip_runtime.h>

// ---------------------------------------------------------------------------
// Types for CDNA5 WMMA
// ---------------------------------------------------------------------------
typedef __bf16 bf16;
typedef __attribute__((ext_vector_type(16))) __bf16 v16bf;
typedef __attribute__((ext_vector_type(8)))  float  v8f;

// Problem constants (match reference)
static constexpr int B_   = 16;
static constexpr int T_   = 96;
static constexpr int N_   = 64;
static constexpr int D_   = 64;
static constexpr int TE_  = 96;
static constexpr int L_   = 2;
static constexpr int F_   = 2;
static constexpr int H_   = 8;
static constexpr int DFF_ = 256;
static constexpr int BN_  = B_ * N_;        // 1024 sequences
static constexpr int TOK_ = BN_ * T_;       // 98304 tokens (also BN_*TE_)

// GEMM flags
#define GF_LOADC 1
#define GF_RELU  2
#define GF_WF    4
#define GF_WH    8
#define GF_BIAS 16

// ---------------------------------------------------------------------------
// Generic WMMA GEMM:  C[M,N] (+)= shift(A)[M,K] @ Bp^T   (Bp stored N-major:
// Bp[n*K + k]).  Each wave computes a 16x64 tile; block = 8 waves = 128 rows.
// shiftMode: 0 = none, 1 = use row m-1 (zero at seq start), 2 = row m+1
// (zero at seq end), sequences are contiguous blocks of `seqLen` rows.
// ---------------------------------------------------------------------------
__global__ __launch_bounds__(256)
void wmma_gemm(const bf16* __restrict__ A, const bf16* __restrict__ Bp,
               const float* __restrict__ bias, float* __restrict__ Cf,
               bf16* __restrict__ Ch, int M, int K, int N,
               int seqLen, int shiftMode, int flags)
{
    const int lane = threadIdx.x & 31;
    const int wave = threadIdx.x >> 5;
    const int g    = lane >> 4;       // half-wave group (0/1)
    const int mr   = lane & 15;       // A-row within tile / B,C-col within tile
    const int tile = blockIdx.x * 8 + wave;
    const int row0 = tile * 16;
    const int colBase = blockIdx.y * 64;

    // Per-lane A source row (with optional conv shift + boundary zeroing)
    int  arow   = row0 + mr;
    bool avalid = true;
    if (shiftMode == 1) { int t = arow % seqLen; avalid = (t > 0);          arow -= 1; }
    else if (shiftMode == 2) { int t = arow % seqLen; avalid = (t < seqLen - 1); arow += 1; }

    v8f acc[4] = {};
    if (flags & GF_LOADC) {
#pragma unroll
        for (int nt = 0; nt < 4; ++nt) {
            const int col = colBase + nt * 16 + mr;
#pragma unroll
            for (int r = 0; r < 8; ++r) {
                const int rr = row0 + r + 8 * g;
                acc[nt][r] = Cf[(size_t)rr * N + col];
            }
        }
    }

    union Frag { uint4 q[2]; v16bf v; };

    for (int kk = 0; kk < K; kk += 32) {
        Frag fa;
        if (avalid) {
            const uint4* ap = reinterpret_cast<const uint4*>(A + (size_t)arow * K + kk);
            fa.q[0] = ap[g];        // elements kk + 8g      .. +7
            fa.q[1] = ap[2 + g];    // elements kk + 16 + 8g .. +7
        } else {
            fa.q[0] = make_uint4(0u, 0u, 0u, 0u);
            fa.q[1] = make_uint4(0u, 0u, 0u, 0u);
        }
#pragma unroll
        for (int nt = 0; nt < 4; ++nt) {
            const int col = colBase + nt * 16 + mr;
            const uint4* bp = reinterpret_cast<const uint4*>(Bp + (size_t)col * K + kk);
            Frag fb;
            fb.q[0] = bp[g];          // elements kk + 8g      .. +7
            fb.q[1] = bp[2 + g];      // elements kk + 16 + 8g .. +7
            acc[nt] = __builtin_amdgcn_wmma_f32_16x16x32_bf16(
                false, fa.v, false, fb.v, (short)0, acc[nt], false, false);
        }
    }

#pragma unroll
    for (int nt = 0; nt < 4; ++nt) {
        const int col = colBase + nt * 16 + mr;
        const float bv = (flags & GF_BIAS) ? bias[col] : 0.f;
#pragma unroll
        for (int r = 0; r < 8; ++r) {
            const int rr = row0 + r + 8 * g;
            float v = acc[nt][r] + bv;
            if (flags & GF_RELU) v = fmaxf(v, 0.f);
            const size_t o = (size_t)rr * N + col;
            if (flags & GF_WF) Cf[o] = v;
            if (flags & GF_WH) Ch[o] = (bf16)v;
        }
    }
}

// ---------------------------------------------------------------------------
// Pack fp32 weight matrix into bf16 "B transposed" layout: dst[n*K + k] =
// src[k*sK + n*sN]  (handles (in,out) weights and conv (out,in,tap) weights)
// ---------------------------------------------------------------------------
__global__ void pack_bt(const float* __restrict__ src, bf16* __restrict__ dst,
                        int K, int N, int sK, int sN)
{
    const int i = blockIdx.x * blockDim.x + threadIdx.x;
    if (i >= K * N) return;
    const int n = i / K, k = i % K;
    dst[(size_t)n * K + k] = (bf16)src[(size_t)k * sK + (size_t)n * sN];
}

__global__ void cast_to_bf16(const float* __restrict__ src, bf16* __restrict__ dst, int n)
{
    const int i = blockIdx.x * blockDim.x + threadIdx.x;
    if (i < n) dst[i] = (bf16)src[i];
}

// ---------------------------------------------------------------------------
// Embedding + positional encoding, with layout change (B,T,N,1)->(BN,T,D)
// ---------------------------------------------------------------------------
__global__ void embed_kernel(const float* __restrict__ x, const float* __restrict__ eW,
                             const float* __restrict__ eb, const float* __restrict__ pe,
                             const int* __restrict__ lk, float* __restrict__ Xf,
                             bf16* __restrict__ Xh)
{
    const int idx = blockIdx.x * blockDim.x + threadIdx.x;   // over TOK_*D_
    if (idx >= TOK_ * D_) return;
    const int d   = idx & 63;
    const int tok = idx >> 6;          // tok = s*T + t
    const int t   = tok % T_;
    const int s   = tok / T_;
    const int b   = s >> 6;            // N_ == 64
    const int n   = s & 63;
    const float xv = x[(size_t)(b * T_ + t) * N_ + n];
    const float v  = xv * eW[d] + eb[d] + pe[(size_t)(*lk + t) * D_ + d];
    Xf[idx] = v;
    Xh[idx] = (bf16)v;
}

// ---------------------------------------------------------------------------
// Multi-head attention core (Dh = 8): one block per sequence, K/V in LDS,
// online softmax in fp32.  causal=1 -> lower-triangular mask.
// ---------------------------------------------------------------------------
__global__ __launch_bounds__(256)
void attn_kernel(const bf16* __restrict__ Q, const bf16* __restrict__ K,
                 const bf16* __restrict__ V, bf16* __restrict__ O,
                 int Tq, int Tk, int causal)
{
    __shared__ float Ks[96 * 64];
    __shared__ float Vs[96 * 64];
    const int seq = blockIdx.x;
    const size_t kbase = (size_t)seq * Tk * D_;
    for (int i = threadIdx.x; i < Tk * D_; i += 256) {
        Ks[i] = (float)K[kbase + i];
        Vs[i] = (float)V[kbase + i];
    }
    __syncthreads();

    const size_t qbase = (size_t)seq * Tq * D_;
    for (int p = threadIdx.x; p < H_ * Tq; p += 256) {
        const int q = p >> 3;
        const int h = p & 7;
        float qv[8];
#pragma unroll
        for (int j = 0; j < 8; ++j) qv[j] = (float)Q[qbase + q * D_ + h * 8 + j];
        const int kmax = causal ? q : (Tk - 1);
        float m = -3.0e38f, l = 0.f;
        float o[8];
#pragma unroll
        for (int j = 0; j < 8; ++j) o[j] = 0.f;
        for (int k = 0; k <= kmax; ++k) {
            const float* kr = &Ks[k * D_ + h * 8];
            float s = 0.f;
#pragma unroll
            for (int j = 0; j < 8; ++j) s += qv[j] * kr[j];
            s *= 0.35355339059327373f;          // 1/sqrt(8)
            const float nm = fmaxf(m, s);
            const float c  = __expf(m - nm);
            const float e  = __expf(s - nm);
            l = l * c + e;
            const float* vr = &Vs[k * D_ + h * 8];
#pragma unroll
            for (int j = 0; j < 8; ++j) o[j] = o[j] * c + e * vr[j];
            m = nm;
        }
        const float inv = 1.f / l;
#pragma unroll
        for (int j = 0; j < 8; ++j)
            O[qbase + q * D_ + h * 8 + j] = (bf16)(o[j] * inv);
    }
}

// ---------------------------------------------------------------------------
// Fused residual + LayerNorm: Y = LN(X + scale*Hd).  One wave per row (D=64),
// lane holds cols {lane, lane+32}; emits fp32 + bf16.
// ---------------------------------------------------------------------------
__global__ __launch_bounds__(256)
void ln_kernel(const float* __restrict__ X, const float* __restrict__ Hd, float scale,
               const float* __restrict__ gam, const float* __restrict__ bet,
               float* __restrict__ Yf, bf16* __restrict__ Yh, int rows)
{
    const int wave = threadIdx.x >> 5;
    const int lane = threadIdx.x & 31;
    const int row  = blockIdx.x * 8 + wave;
    if (row >= rows) return;
    const size_t base = (size_t)row * D_;
    const float x0 = X[base + lane]      + scale * Hd[base + lane];
    const float x1 = X[base + 32 + lane] + scale * Hd[base + 32 + lane];
    float s  = x0 + x1;
    float s2 = x0 * x0 + x1 * x1;
#pragma unroll
    for (int off = 16; off >= 1; off >>= 1) {
        s  += __shfl_xor(s,  off, 32);
        s2 += __shfl_xor(s2, off, 32);
    }
    const float mean = s * (1.f / 64.f);
    const float var  = s2 * (1.f / 64.f) - mean * mean;
    const float inv  = rsqrtf(var + 1e-5f);
    const float y0 = (x0 - mean) * inv * gam[lane]      + bet[lane];
    const float y1 = (x1 - mean) * inv * gam[32 + lane] + bet[32 + lane];
    Yf[base + lane]      = y0;  Yf[base + 32 + lane] = y1;
    Yh[base + lane]      = (bf16)y0;
    Yh[base + 32 + lane] = (bf16)y1;
}

// ---------------------------------------------------------------------------
// Final head: y = X @ fc_W + fc_b  (D->1), with (BN,T)->(B,T,N) transpose
// ---------------------------------------------------------------------------
__global__ void head_kernel(const float* __restrict__ X, const float* __restrict__ W,
                            const float* __restrict__ bb, float* __restrict__ out)
{
    const int tok = blockIdx.x * blockDim.x + threadIdx.x;   // tok = s*T + t
    if (tok >= TOK_) return;
    const float* xr = X + (size_t)tok * D_;
    float acc = 0.f;
#pragma unroll
    for (int d = 0; d < D_; ++d) acc += xr[d] * W[d];
    acc += bb[0];
    const int t = tok % T_, s = tok / T_;
    const int b = s >> 6, n = s & 63;
    out[(size_t)(b * T_ + t) * N_ + n] = acc;
}

// ---------------------------------------------------------------------------
// Host orchestration
// ---------------------------------------------------------------------------
extern "C" void kernel_launch(void* const* d_in, const int* in_sizes, int n_in,
                              void* d_out, int out_size, void* d_ws, size_t ws_size,
                              hipStream_t stream)
{
    (void)in_sizes; (void)n_in; (void)out_size; (void)ws_size;

    const float* x      = (const float*)d_in[0];
    const float* enc[2] = {(const float*)d_in[1], (const float*)d_in[2]};
    // d_in[3] = tgt_mask (causal; implied in attention kernel)
    const float* embW   = (const float*)d_in[4];
    const float* embB   = (const float*)d_in[5];
    const float* pe     = (const float*)d_in[6];
    const float* convqW = (const float*)d_in[7];
    const float* convkW = (const float*)d_in[8];
    const float* convkB = (const float*)d_in[9];
    const float* saW    = (const float*)d_in[10];
    const float* saB    = (const float*)d_in[11];
    const float* caW    = (const float*)d_in[12];
    const float* caB    = (const float*)d_in[13];
    const float* lnG    = (const float*)d_in[14];
    const float* lnB    = (const float*)d_in[15];
    const float* f1W    = (const float*)d_in[16];
    const float* f1B    = (const float*)d_in[17];
    const float* f2W    = (const float*)d_in[18];
    const float* f2B    = (const float*)d_in[19];
    const float* fcW    = (const float*)d_in[20];
    const float* fcB    = (const float*)d_in[21];
    const int*   lk     = (const int*)d_in[22];
    float* out          = (float*)d_out;

    // -------- workspace carve-up --------
    char* wp = (char*)d_ws;
    auto alloc = [&](size_t bytes) -> void* {
        void* p = (void*)wp;
        wp += (bytes + 255) & ~(size_t)255;
        return p;
    };
    const size_t AF = (size_t)TOK_ * D_ * sizeof(float);   // fp32 activation
    const size_t AH = (size_t)TOK_ * D_ * sizeof(bf16);    // bf16 activation

    float* X0f = (float*)alloc(AF);  bf16* X0h = (bf16*)alloc(AH);
    float* X1f = (float*)alloc(AF);  bf16* X1h = (bf16*)alloc(AH);
    bf16*  qH  = (bf16*)alloc(AH);
    bf16*  kH  = (bf16*)alloc(AH);
    bf16*  vH  = (bf16*)alloc(AH);
    bf16*  oH  = (bf16*)alloc(AH);
    bf16*  eH  = (bf16*)alloc(AH);                         // enc conv output (bf16)
    bf16*  encH[2] = {(bf16*)alloc(AH), (bf16*)alloc(AH)}; // enc inputs, bf16
    float* tF  = (float*)alloc(AF);                        // generic fp32 GEMM temp
    float* cF  = (float*)alloc(AF);                        // cross-attn accumulator
    bf16*  hH  = (bf16*)alloc((size_t)TOK_ * DFF_ * sizeof(bf16)); // ffn hidden

    bf16* trendBp[L_][2];
    bf16* convkBp[L_][F_][3];
    bf16* saBp[L_][4];
    bf16* caBp[L_][F_][4];
    bf16* f1Bp[L_];
    bf16* f2Bp[L_];
    for (int l = 0; l < L_; ++l) {
        for (int tp = 0; tp < 2; ++tp) trendBp[l][tp] = (bf16*)alloc(64 * 64 * 2);
        for (int f = 0; f < F_; ++f)
            for (int tp = 0; tp < 3; ++tp) convkBp[l][f][tp] = (bf16*)alloc(64 * 64 * 2);
        for (int j = 0; j < 4; ++j) saBp[l][j] = (bf16*)alloc(64 * 64 * 2);
        for (int f = 0; f < F_; ++f)
            for (int j = 0; j < 4; ++j) caBp[l][f][j] = (bf16*)alloc(64 * 64 * 2);
        f1Bp[l] = (bf16*)alloc(64 * 256 * 2);
        f2Bp[l] = (bf16*)alloc(256 * 64 * 2);
    }

    // -------- launch helpers --------
    auto pack = [&](const float* src, bf16* dst, int K, int N, int sK, int sN) {
        const int tot = K * N;
        pack_bt<<<(tot + 255) / 256, 256, 0, stream>>>(src, dst, K, N, sK, sN);
    };
    auto gemm = [&](const bf16* A, const bf16* Bp, const float* bias, float* Cf,
                    bf16* Ch, int M, int K, int N, int shift, int flags) {
        dim3 grid(M / 128, N / 64);
        wmma_gemm<<<grid, 256, 0, stream>>>(A, Bp, bias, Cf, Ch, M, K, N,
                                            /*seqLen=*/96, shift, flags);
    };

    // -------- weight packing (fp32 -> bf16, N-major "B^T" layout) --------
    for (int l = 0; l < L_; ++l) {
        // convq_W: (L, Dout, Din, 3); B(k=i, n=o) = W[o,i,tap]
        for (int tp = 0; tp < 2; ++tp)
            pack(convqW + (size_t)l * 64 * 64 * 3 + tp, trendBp[l][tp], 64, 64, 3, 192);
        // convk_W: (L, F, Dout, Din, 3)
        for (int f = 0; f < F_; ++f)
            for (int tp = 0; tp < 3; ++tp)
                pack(convkW + ((size_t)(l * F_ + f)) * 64 * 64 * 3 + tp,
                     convkBp[l][f][tp], 64, 64, 3, 192);
        // sa_W: (L, 4, D, D) stored (in, out)
        for (int j = 0; j < 4; ++j)
            pack(saW + ((size_t)(l * 4 + j)) * 64 * 64, saBp[l][j], 64, 64, 64, 1);
        // ca_W: (L, F, 4, D, D) stored (in, out)
        for (int f = 0; f < F_; ++f)
            for (int j = 0; j < 4; ++j)
                pack(caW + ((size_t)((l * F_ + f) * 4 + j)) * 64 * 64,
                     caBp[l][f][j], 64, 64, 64, 1);
        pack(f1W + (size_t)l * 64 * 256, f1Bp[l], 64, 256, 256, 1);
        pack(f2W + (size_t)l * 256 * 64, f2Bp[l], 256, 64, 64, 1);
    }

    // -------- encoder inputs to bf16 --------
    for (int f = 0; f < F_; ++f)
        cast_to_bf16<<<(TOK_ * D_ + 255) / 256, 256, 0, stream>>>(enc[f], encH[f], TOK_ * D_);

    // -------- embedding + positional --------
    embed_kernel<<<(TOK_ * D_ + 255) / 256, 256, 0, stream>>>(x, embW, embB, pe, lk, X0f, X0h);

    float* Xf = X0f; bf16* Xh = X0h;
    float* Yf = X1f; bf16* Yh = X1h;
    auto swapXY = [&]() {
        float* tf = Xf; Xf = Yf; Yf = tf;
        bf16*  th = Xh; Xh = Yh; Yh = th;
    };

    for (int l = 0; l < L_; ++l) {
        // ---- local trend (replaces out): Y = xprev @ W0^T + x @ W1^T ----
        gemm(Xh, trendBp[l][0], nullptr, Yf, nullptr, TOK_, 64, 64, 1, GF_WF);
        gemm(Xh, trendBp[l][1], nullptr, Yf, Yh,     TOK_, 64, 64, 0,
             GF_LOADC | GF_WF | GF_WH);
        swapXY();

        // ---- masked self-attention ----
        gemm(Xh, saBp[l][0], saB + (size_t)(l * 4 + 0) * 64, nullptr, qH,
             TOK_, 64, 64, 0, GF_BIAS | GF_WH);
        gemm(Xh, saBp[l][1], saB + (size_t)(l * 4 + 1) * 64, nullptr, kH,
             TOK_, 64, 64, 0, GF_BIAS | GF_WH);
        gemm(Xh, saBp[l][2], saB + (size_t)(l * 4 + 2) * 64, nullptr, vH,
             TOK_, 64, 64, 0, GF_BIAS | GF_WH);
        attn_kernel<<<BN_, 256, 0, stream>>>(qH, kH, vH, oH, T_, T_, 1);
        gemm(oH, saBp[l][3], saB + (size_t)(l * 4 + 3) * 64, tF, nullptr,
             TOK_, 64, 64, 0, GF_BIAS | GF_WF);
        ln_kernel<<<TOK_ / 8, 256, 0, stream>>>(Xf, tF, 1.0f,
            lnG + (size_t)(l * 3 + 0) * 64, lnB + (size_t)(l * 3 + 0) * 64,
            Yf, Yh, TOK_);
        swapXY();

        // ---- cross-attention over F encoder features (mean) ----
        for (int f = 0; f < F_; ++f) {
            // enc conv1d SAME (3 taps) -> eH (bf16); fp32 accumulation in tF
            gemm(encH[f], convkBp[l][f][0], nullptr, tF, nullptr, TOK_, 64, 64, 1, GF_WF);
            gemm(encH[f], convkBp[l][f][1], nullptr, tF, nullptr, TOK_, 64, 64, 0,
                 GF_LOADC | GF_WF);
            gemm(encH[f], convkBp[l][f][2], convkB + (size_t)(l * F_ + f) * 64,
                 tF, eH, TOK_, 64, 64, 2, GF_LOADC | GF_BIAS | GF_WH);
            // Q from decoder state, K/V from conv'd encoder
            const float* cb = caB + (size_t)((l * F_ + f) * 4) * 64;
            gemm(Xh, caBp[l][f][0], cb + 0 * 64, nullptr, qH, TOK_, 64, 64, 0,
                 GF_BIAS | GF_WH);
            gemm(eH, caBp[l][f][1], cb + 1 * 64, nullptr, kH, TOK_, 64, 64, 0,
                 GF_BIAS | GF_WH);
            gemm(eH, caBp[l][f][2], cb + 2 * 64, nullptr, vH, TOK_, 64, 64, 0,
                 GF_BIAS | GF_WH);
            attn_kernel<<<BN_, 256, 0, stream>>>(qH, kH, vH, oH, T_, TE_, 0);
            gemm(oH, caBp[l][f][3], cb + 3 * 64, cF, nullptr, TOK_, 64, 64, 0,
                 GF_BIAS | GF_WF | (f ? GF_LOADC : 0));
        }
        ln_kernel<<<TOK_ / 8, 256, 0, stream>>>(Xf, cF, 0.5f,
            lnG + (size_t)(l * 3 + 1) * 64, lnB + (size_t)(l * 3 + 1) * 64,
            Yf, Yh, TOK_);
        swapXY();

        // ---- FFN ----
        gemm(Xh, f1Bp[l], f1B + (size_t)l * 256, nullptr, hH, TOK_, 64, 256, 0,
             GF_BIAS | GF_RELU | GF_WH);
        gemm(hH, f2Bp[l], f2B + (size_t)l * 64, tF, nullptr, TOK_, 256, 64, 0,
             GF_BIAS | GF_WF);
        ln_kernel<<<TOK_ / 8, 256, 0, stream>>>(Xf, tF, 1.0f,
            lnG + (size_t)(l * 3 + 2) * 64, lnB + (size_t)(l * 3 + 2) * 64,
            Yf, Yh, TOK_);
        swapXY();
    }

    // ---- final projection + transpose to (B,T,N,1) ----
    head_kernel<<<TOK_ / 256, 256, 0, stream>>>(Xf, fcW, fcB, out);
}
